// GeneExpressionTokeniser_20693152432936
// MI455X (gfx1250) — matrix operand order repaired and checked by reading, not compiled, not run on hardware
//
#include <hip/hip_runtime.h>
#include <hip/hip_bf16.h>
#include <math.h>

#define BB  16
#define GG  20000
#define DD  256
#define KK  2048
#define KP1 2049

typedef _Float16 v16h __attribute__((ext_vector_type(16)));
typedef _Float16 v8h  __attribute__((ext_vector_type(8)));
typedef float    v8f  __attribute__((ext_vector_type(8)));

// ---------------------------------------------------------------------------
// Kernel 1: convert w2 (f32, row-major [D][D]) to f16 row-major.
// B-matrix column e == w2 row e, so fragments load contiguously from this.
// ---------------------------------------------------------------------------
__global__ void convert_w2(const float* __restrict__ w2, _Float16* __restrict__ w2h) {
    int i = blockIdx.x * blockDim.x + threadIdx.x;
    if (i < DD * DD) w2h[i] = (_Float16)w2[i];
}

// ---------------------------------------------------------------------------
// Kernel 2: per-batch-row radix select of the Kth largest expression value.
// Positive floats order identically to their uint bit patterns, so 4 passes
// of 8-bit digit histograms give the exact threshold plus the tie rank
// (= how many values equal to the threshold are kept, lowest index first,
// matching jax.lax.top_k's stable tie behaviour).
// ---------------------------------------------------------------------------
__global__ void select_thresh(const float* __restrict__ expr,
                              unsigned* __restrict__ tBits,
                              int* __restrict__ needOut) {
    int b = blockIdx.x;
    __shared__ int hist[256];
    __shared__ unsigned sPrefix;
    __shared__ int sNeed;
    if (threadIdx.x == 0) { sPrefix = 0u; sNeed = KK; }
    for (int pass = 0; pass < 4; ++pass) {
        int shift = 24 - pass * 8;
        for (int i = threadIdx.x; i < 256; i += blockDim.x) hist[i] = 0;
        __syncthreads();
        unsigned prefix = sPrefix;
        unsigned maskHi = (pass == 0) ? 0u : (0xFFFFFFFFu << (shift + 8));
        for (int g = threadIdx.x; g < GG; g += blockDim.x) {
            unsigned u = __float_as_uint(expr[b * GG + g]);
            if ((u & maskHi) == prefix)
                atomicAdd(&hist[(u >> shift) & 0xFF], 1);
        }
        __syncthreads();
        if (threadIdx.x == 0) {
            int need = sNeed, cum = 0, chosen = 0;
            for (int d = 255; d >= 0; --d) {
                cum += hist[d];
                if (cum >= need) { chosen = d; sNeed = need - (cum - hist[d]); break; }
            }
            sPrefix = prefix | ((unsigned)chosen << shift);
        }
        __syncthreads();
    }
    if (threadIdx.x == 0) { tBits[b] = sPrefix; needOut[b] = sNeed; }
}

// ---------------------------------------------------------------------------
// Kernel 3: stable stream compaction. Keep genes with expr > t, plus the
// first `need` genes (ascending index) with expr == t; all must be > 0.
// Produces ascending compacted indices (pad = -1) and the float mask row.
// ---------------------------------------------------------------------------
__global__ void compact_kernel(const float* __restrict__ expr,
                               const unsigned* __restrict__ tBits,
                               const int* __restrict__ needArr,
                               int* __restrict__ compIdx,
                               float* __restrict__ maskOut) {
    int b = blockIdx.x;
    float t = __uint_as_float(tBits[b]);
    int need = needArr[b];
    __shared__ int warpEq[8], warpKeep[8];
    __shared__ int sBase, sTies;
    if (threadIdx.x == 0) { sBase = 0; sTies = 0; }
    __syncthreads();
    int lane = threadIdx.x & 31;
    int warp = threadIdx.x >> 5;
    unsigned long long laneMask = (1ull << lane) - 1ull;

    for (int chunk = 0; chunk < GG; chunk += 256) {
        int g = chunk + threadIdx.x;
        float v = (g < GG) ? expr[b * GG + g] : -1.0f;
        bool isGt = (v > t) && (v > 0.0f);
        bool isEq = (v == t) && (v > 0.0f);
        unsigned long long mEq = __ballot(isEq);
        if (lane == 0) warpEq[warp] = __popcll(mEq);
        __syncthreads();
        int eqOff = 0, eqTot = 0;
        #pragma unroll
        for (int w = 0; w < 8; ++w) { if (w < warp) eqOff += warpEq[w]; eqTot += warpEq[w]; }
        int tieRank = sTies + eqOff + __popcll(mEq & laneMask);
        bool keep = isGt || (isEq && tieRank < need);
        unsigned long long mKeep = __ballot(keep);
        if (lane == 0) warpKeep[warp] = __popcll(mKeep);
        __syncthreads();
        int kOff = 0, kTot = 0;
        #pragma unroll
        for (int w = 0; w < 8; ++w) { if (w < warp) kOff += warpKeep[w]; kTot += warpKeep[w]; }
        int pos = sBase + kOff + __popcll(mKeep & laneMask);
        if (keep && pos < KK) compIdx[b * KK + pos] = g;
        __syncthreads();
        if (threadIdx.x == 0) { sBase += kTot; sTies += eqTot; }
        __syncthreads();
    }
    int kept = sBase;
    for (int s = kept + (int)threadIdx.x; s < KK; s += blockDim.x)
        compIdx[b * KK + s] = -1;
    if (threadIdx.x == 0) maskOut[(size_t)b * KP1] = 1.0f;
    for (int s = threadIdx.x; s < KK; s += blockDim.x)
        maskOut[(size_t)b * KP1 + 1 + s] = (s < kept) ? 1.0f : 0.0f;
}

// ---------------------------------------------------------------------------
// Kernel 4: CLS token rows.
// ---------------------------------------------------------------------------
__global__ void cls_fill(const float* __restrict__ cls, float* __restrict__ out) {
    int b = blockIdx.x;
    int d = threadIdx.x;
    out[(size_t)b * KP1 * DD + d] = cls[d];
}

// ---------------------------------------------------------------------------
// Kernel 5: fused gather + MLP GEMM with f16 WMMA.
// Block = 128 threads (4 wave32s), owns 16 tokens. Stage h = gelu(x*w1+b1)
// as f16 in LDS (row pad 8 halves -> 4-bank row skew, conflict-free b128
// A-fragment loads). Each wave computes a 16x64 slab of val_emb via
// v_wmma_f32_16x16x32_f16, K-loop over 256. Epilogue adds b2 + gene_emb
// gather, zeroes invalid slots.
// ---------------------------------------------------------------------------
__global__ __launch_bounds__(128)
void token_gemm(const float* __restrict__ expr,
                const float* __restrict__ gene_emb,
                const float* __restrict__ w1,
                const float* __restrict__ b1,
                const _Float16* __restrict__ w2h,
                const float* __restrict__ b2,
                const int* __restrict__ compIdx,
                float* __restrict__ out) {
    __shared__ _Float16 hA[16][DD + 8];
    __shared__ int gIdx[16];
    __shared__ float xV[16];
    __shared__ float wS[DD], bS[DD];

    int bIdx = blockIdx.x >> 7;              // KK/16 = 128 M-tiles per batch
    int slotBase = (blockIdx.x & 127) << 4;

    for (int i = threadIdx.x; i < DD; i += 128) { wS[i] = w1[i]; bS[i] = b1[i]; }
    if (threadIdx.x < 16) {
        int g = compIdx[bIdx * KK + slotBase + threadIdx.x];
        gIdx[threadIdx.x] = g;
        xV[threadIdx.x] = (g >= 0) ? expr[bIdx * GG + g] : 0.0f;
    }
    __syncthreads();

    // Stage h (exact GELU) as f16, ISA 16-bit A layout handled at load time.
    for (int i = threadIdx.x; i < 16 * DD; i += 128) {
        int m = i >> 8;
        int d = i & 255;
        float hv = 0.0f;
        if (gIdx[m] >= 0) {
            float z = xV[m] * wS[d] + bS[d];
            hv = 0.5f * z * (1.0f + erff(z * 0.70710678118654752f));
        }
        hA[m][d] = (_Float16)hv;
    }
    __syncthreads();

    int lane = threadIdx.x & 31;
    int wv   = threadIdx.x >> 5;
    int hi   = lane >> 4;        // half-wave select
    int rowM = lane & 15;
    int nBase = wv * 64;

    v8f acc[4] = {v8f{}, v8f{}, v8f{}, v8f{}};
    const _Float16* hrow = &hA[rowM][0];

    for (int k0 = 0; k0 < DD; k0 += 32) {
        // A fragment (16x32 f16): lanes 0-15 hold K {0-15}, lanes 16-31 K {16-31}
        v8h alo = *(const v8h*)(hrow + k0 + hi * 16);
        v8h ahi = *(const v8h*)(hrow + k0 + hi * 16 + 8);
        v16h a = __builtin_shufflevector(alo, ahi,
                                         0, 1, 2, 3, 4, 5, 6, 7,
                                         8, 9, 10, 11, 12, 13, 14, 15);
        int koff = k0 + hi * 16; // B: lane = column N, K sequential per half-wave
        #pragma unroll
        for (int nt = 0; nt < 4; ++nt) {
            int col = nBase + nt * 16 + rowM;
            v16h bf = *(const v16h*)(w2h + (size_t)col * DD + koff);
            acc[nt] = __builtin_amdgcn_wmma_f32_16x16x32_f16(
                false, a, false, bf, (short)0, acc[nt], false, false);
        }
    }

    // Epilogue: C layout VGPR r -> (M=r | r+8, N=lane&15). Fuse b2 + gene_emb.
    float* outB = out + (size_t)bIdx * KP1 * DD;
    #pragma unroll
    for (int nt = 0; nt < 4; ++nt) {
        int n = nBase + nt * 16 + rowM;
        float bias = b2[n];
        #pragma unroll
        for (int r = 0; r < 8; ++r) {
            int m = r + hi * 8;
            int g = gIdx[m];
            float val = 0.0f;
            if (g >= 0) val = acc[nt][r] + bias + gene_emb[(size_t)g * DD + n];
            outB[(size_t)(slotBase + m + 1) * DD + n] = val;
        }
    }
}

// ---------------------------------------------------------------------------
extern "C" void kernel_launch(void* const* d_in, const int* in_sizes, int n_in,
                              void* d_out, int out_size, void* d_ws, size_t ws_size,
                              hipStream_t stream) {
    const float* expr     = (const float*)d_in[0];
    const float* gene_emb = (const float*)d_in[1];
    const float* w1       = (const float*)d_in[2];
    const float* b1       = (const float*)d_in[3];
    const float* w2       = (const float*)d_in[4];
    const float* b2       = (const float*)d_in[5];
    const float* cls      = (const float*)d_in[6];
    (void)in_sizes; (void)n_in; (void)out_size; (void)ws_size;

    float* out = (float*)d_out;
    char* ws = (char*)d_ws;
    _Float16* w2h    = (_Float16*)ws;                 // 256*256*2   = 131072 B
    int*      cIdx   = (int*)(ws + 131072);           // 16*2048*4   = 131072 B
    unsigned* tBits  = (unsigned*)(ws + 262144);      // 64 B
    int*      needA  = (int*)(ws + 262208);           // 64 B
    float*    maskP  = out + (size_t)BB * KP1 * DD;   // mask region of d_out

    hipLaunchKernelGGL(convert_w2,    dim3((DD * DD) / 256), dim3(256), 0, stream, w2, w2h);
    hipLaunchKernelGGL(select_thresh, dim3(BB),              dim3(256), 0, stream, expr, tBits, needA);
    hipLaunchKernelGGL(compact_kernel,dim3(BB),              dim3(256), 0, stream, expr, tBits, needA, cIdx, maskP);
    hipLaunchKernelGGL(cls_fill,      dim3(BB),              dim3(DD),  0, stream, cls, out);
    hipLaunchKernelGGL(token_gemm,    dim3(BB * KK / 16),    dim3(128), 0, stream,
                       expr, gene_emb, w1, b1, w2h, b2, cIdx, out);
}
